// SimpleGCN_2310692405528
// MI455X (gfx1250) — matrix operand order repaired and checked by reading, not compiled, not run on hardware
//
#include <hip/hip_runtime.h>
#include <hip/hip_bf16.h>
#include <cstddef>

typedef __attribute__((ext_vector_type(16))) _Float16 v16h;
typedef __attribute__((ext_vector_type(8)))  _Float16 v8h;
typedef __attribute__((ext_vector_type(8)))  float    v8f;

// ---------------------------------------------------------------------------
// Small utility kernels
// ---------------------------------------------------------------------------
__global__ void fill_f32(float* p, float v, int n) {
    int t = blockIdx.x * blockDim.x + threadIdx.x;
    if (t < n) p[t] = v;
}

__global__ void deg_accum(const int* __restrict__ dst, float* __restrict__ deg, int nE) {
    int t = blockIdx.x * blockDim.x + threadIdx.x;
    if (t < nE) atomicAdd(&deg[dst[t]], 1.0f);
}

__global__ void rsqrt_inplace(float* p, int n) {
    int t = blockIdx.x * blockDim.x + threadIdx.x;
    if (t < n) p[t] = rsqrtf(p[t]);
}

__global__ void cast_f32_to_f16(const float* __restrict__ s, _Float16* __restrict__ d, int n) {
    int t = blockIdx.x * blockDim.x + threadIdx.x;
    if (t < n) d[t] = (_Float16)s[t];
}

// Wt[n*K + k] = (f16) W[k*Nc + n]   (K = input dim = 64, Nc = output dim)
__global__ void transpose_cast(const float* __restrict__ W, _Float16* __restrict__ Wt,
                               int K, int Nc) {
    int t = blockIdx.x * blockDim.x + threadIdx.x;
    if (t < K * Nc) {
        int n = t / K, k = t % K;
        Wt[n * K + k] = (_Float16)W[(size_t)k * Nc + n];
    }
}

// ---------------------------------------------------------------------------
// WMMA GEMM:  H[nrows, COUT] = Ah[nrows, 64] * Bt[COUT, 64]^T   (f16 in, f32 out)
// Bt is W^T stored row-major ([COUT][64]) so each lane's B fragment is a
// contiguous 32-byte load.  One wave per 16-row M tile; 4 waves per block.
// ---------------------------------------------------------------------------
template <int COUT>
__launch_bounds__(128)
__global__ void gemm_wmma_f16(const _Float16* __restrict__ Ah,
                              const _Float16* __restrict__ Bt,
                              float* __restrict__ H, int nrows) {
    constexpr int KDIM = 64;
    constexpr int NT   = COUT / 16;

    const int lane = threadIdx.x & 31;
    const int wave = threadIdx.x >> 5;
    const int row0 = blockIdx.x * 64 + wave * 16;
    if (row0 >= nrows) return;  // whole-wave uniform exit

    // A fragment addressing (contiguous k = 16h + s order used by this code):
    //   lane<16 : M=row0+lane,     K = 0..15  (+k0)
    //   lane>=16: M=row0+lane-16,  K = 16..31 (+k0)
    int m = row0 + (lane & 15);
    m = min(m, nrows - 1);                 // clamp tail rows (EXEC must stay all-1)
    const int khalf = (lane >> 4) * 16;    // 0 or 16
    const _Float16* ap = Ah + (size_t)m * KDIM;

    // B fragment addressing (same contiguous k order as A):
    //   lane<16 : N=n0+lane,    K = 0..15  (+k0)
    //   lane>=16: N=n0+lane-16, K = 16..31 (+k0)
    const int koff = (lane >> 4) * 16;     // 0 or 16
    const int ncol = lane & 15;

    v8f acc[NT];
#pragma unroll
    for (int nt = 0; nt < NT; ++nt) acc[nt] = (v8f)(0.0f);

#pragma unroll
    for (int k0 = 0; k0 < KDIM; k0 += 32) {
        v8h lo = *(const v8h*)(ap + k0 + khalf);
        v8h hi = *(const v8h*)(ap + k0 + 8 + khalf);
        v16h a;
#pragma unroll
        for (int i = 0; i < 8; ++i) { a[i] = lo[i]; a[i + 8] = hi[i]; }

#pragma unroll
        for (int nt = 0; nt < NT; ++nt) {
            const int n = nt * 16 + ncol;
            v16h b = *(const v16h*)(Bt + (size_t)n * KDIM + k0 + koff);
            acc[nt] = __builtin_amdgcn_wmma_f32_16x16x32_f16(
                false, a, false, b, (short)0, acc[nt], false, false);
        }
    }

    // C/D layout: VGPR r -> lanes 0-15: (M=r, N=lane); lanes 16-31: (M=r+8, N=lane-16)
    const int rbase = row0 + ((lane >> 4) << 3);
    float* hp = H + (size_t)rbase * COUT + ncol;
    if (row0 + 16 <= nrows) {
        // fast path: full tile, no predication (all but the last block)
#pragma unroll
        for (int nt = 0; nt < NT; ++nt)
#pragma unroll
            for (int r = 0; r < 8; ++r)
                hp[(size_t)r * COUT + nt * 16] = acc[nt][r];
    } else {
#pragma unroll
        for (int nt = 0; nt < NT; ++nt)
#pragma unroll
            for (int r = 0; r < 8; ++r)
                if (rbase + r < nrows) hp[(size_t)r * COUT + nt * 16] = acc[nt][r];
    }
}

// ---------------------------------------------------------------------------
// Edge scatter:  agg[dst] += h[src] * dinv[src]*dinv[dst]
// One thread per (edge, 4-channel group); float4 gather, 4 f32 atomics (in L2).
// ---------------------------------------------------------------------------
template <int C>
__global__ void scatter_edges(const float* __restrict__ H,
                              const int* __restrict__ src,
                              const int* __restrict__ dst,
                              const float* __restrict__ dinv,
                              float* __restrict__ agg, int nE) {
    constexpr int G = C / 4;
    int t = blockIdx.x * blockDim.x + threadIdx.x;
    int e = t / G;
    if (e >= nE) return;
    int c4 = (t - e * G) * 4;
    int s = src[e], d = dst[e];
    float w = dinv[s] * dinv[d];
    const float4 v = *reinterpret_cast<const float4*>(H + (size_t)s * C + c4);
    float* ap = agg + (size_t)d * C + c4;
    atomicAdd(ap + 0, v.x * w);
    atomicAdd(ap + 1, v.y * w);
    atomicAdd(ap + 2, v.z * w);
    atomicAdd(ap + 3, v.w * w);
}

// Layer-1 epilogue fused: r1h = (f16) relu(agg + h*dinv^2 + b)
__global__ void self_bias_relu_cast(const float* __restrict__ agg,
                                    const float* __restrict__ H,
                                    const float* __restrict__ dinv,
                                    const float* __restrict__ b,
                                    _Float16* __restrict__ outh, int n) {
    constexpr int C = 64;
    int t = blockIdx.x * blockDim.x + threadIdx.x;
    if (t >= n * C) return;
    int i = t / C, c = t - i * C;
    float di = dinv[i];
    float v = agg[t] + H[t] * di * di + b[c];
    outh[t] = (_Float16)fmaxf(v, 0.0f);
}

// Layer-2 epilogue fused with global mean-pool accumulation:
//   v = agg + h*dinv^2 + b ;  sums[batch[i]][c] += v
__global__ void self_bias_pool(const float* __restrict__ agg,
                               const float* __restrict__ H,
                               const float* __restrict__ dinv,
                               const float* __restrict__ b,
                               const int* __restrict__ batch,
                               float* __restrict__ sums,
                               float* __restrict__ counts, int n) {
    constexpr int C = 32;
    int t = blockIdx.x * blockDim.x + threadIdx.x;
    if (t >= n * C) return;
    int i = t >> 5, c = t & 31;
    float di = dinv[i];
    float v = agg[t] + H[t] * di * di + b[c];
    int g = batch[i];
    atomicAdd(&sums[g * C + c], v);
    if (c == 0) atomicAdd(&counts[g], 1.0f);
}

__global__ void pool_final(const float* __restrict__ sums,
                           const float* __restrict__ counts,
                           float* __restrict__ out, int nG) {
    int t = blockIdx.x * blockDim.x + threadIdx.x;
    if (t >= nG * 32) return;
    out[t] = sums[t] / fmaxf(counts[t >> 5], 1.0f);
}

// ---------------------------------------------------------------------------
// Launch
// ---------------------------------------------------------------------------
static inline int cdiv(long a, int b) { return (int)((a + b - 1) / b); }

extern "C" void kernel_launch(void* const* d_in, const int* in_sizes, int n_in,
                              void* d_out, int out_size, void* d_ws, size_t ws_size,
                              hipStream_t stream) {
    const float* x    = (const float*)d_in[0];
    const int*   ei   = (const int*)d_in[1];
    const int*   batch= (const int*)d_in[2];
    const float* W1   = (const float*)d_in[3];
    const float* b1   = (const float*)d_in[4];
    const float* W2   = (const float*)d_in[5];
    const float* b2   = (const float*)d_in[6];
    float* out = (float*)d_out;

    const int IN_CH = 64, HID = 64, OUT = 32;
    const int N  = in_sizes[0] / IN_CH;
    const int E  = in_sizes[1] / 2;
    const int G  = out_size / OUT;
    const int* src = ei;
    const int* dst = ei + E;

    // Workspace carve-out (256B aligned)
    char* base = (char*)d_ws;
    size_t off = 0;
    auto alloc = [&](size_t bytes) -> char* {
        char* p = base + off;
        off = (off + bytes + 255) & ~(size_t)255;
        return p;
    };
    float*    dinv  = (float*)   alloc((size_t)N * 4);
    _Float16* xh    = (_Float16*)alloc((size_t)N * HID * 2);   // GEMM1 input, reused as GEMM2 input
    _Float16* w1t   = (_Float16*)alloc((size_t)HID * IN_CH * 2);
    _Float16* w2t   = (_Float16*)alloc((size_t)OUT * HID * 2);
    float*    h1    = (float*)   alloc((size_t)N * HID * 4);
    float*    agg1  = (float*)   alloc((size_t)N * HID * 4);
    float*    h2    = (float*)   alloc((size_t)N * OUT * 4);
    float*    agg2  = (float*)   alloc((size_t)N * OUT * 4);
    float*    sums  = (float*)   alloc((size_t)G * OUT * 4);
    float*    cnts  = (float*)   alloc((size_t)G * 4);
    (void)ws_size; (void)n_in;

    const int B = 256;

    // --- degree with self loops -> dinv ---
    fill_f32<<<cdiv(N, B), B, 0, stream>>>(dinv, 1.0f, N);
    deg_accum<<<cdiv(E, B), B, 0, stream>>>(dst, dinv, E);
    rsqrt_inplace<<<cdiv(N, B), B, 0, stream>>>(dinv, N);

    // --- f16 conversions / weight transpose ---
    cast_f32_to_f16<<<cdiv((long)N * IN_CH, B), B, 0, stream>>>(x, xh, N * IN_CH);
    transpose_cast<<<cdiv(IN_CH * HID, B), B, 0, stream>>>(W1, w1t, IN_CH, HID);
    transpose_cast<<<cdiv(HID * OUT, B), B, 0, stream>>>(W2, w2t, HID, OUT);

    // --- layer 1: h1 = x @ W1 (WMMA), scatter, fused self+bias+relu -> f16 ---
    gemm_wmma_f16<64><<<cdiv(N, 64), 128, 0, stream>>>(xh, w1t, h1, N);
    fill_f32<<<cdiv((long)N * HID, B), B, 0, stream>>>(agg1, 0.0f, N * HID);
    scatter_edges<64><<<cdiv((long)E * (HID / 4), B), B, 0, stream>>>(h1, src, dst, dinv, agg1, E);
    self_bias_relu_cast<<<cdiv((long)N * HID, B), B, 0, stream>>>(agg1, h1, dinv, b1, xh, N);

    // --- layer 2: h2 = r1 @ W2 (WMMA), scatter, fused self+bias+pool ---
    gemm_wmma_f16<32><<<cdiv(N, 64), 128, 0, stream>>>(xh, w2t, h2, N);
    fill_f32<<<cdiv((long)N * OUT, B), B, 0, stream>>>(agg2, 0.0f, N * OUT);
    scatter_edges<32><<<cdiv((long)E * (OUT / 4), B), B, 0, stream>>>(h2, src, dst, dinv, agg2, E);

    fill_f32<<<cdiv(G * OUT + G, B), B, 0, stream>>>(sums, 0.0f, G * OUT + G); // sums + cnts contiguous
    self_bias_pool<<<cdiv((long)N * OUT, B), B, 0, stream>>>(agg2, h2, dinv, b2, batch, sums, cnts, N);
    pool_final<<<cdiv(G * 32, B), B, 0, stream>>>(sums, cnts, out, G);
}
